// EncoderBlock_59150289600833
// MI455X (gfx1250) — matrix operand order repaired and checked by reading, not compiled, not run on hardware
//
#include <hip/hip_runtime.h>
#include <hip/hip_bf16.h>
#include <math.h>

// ---------------------------------------------------------------------------
// Types matching the gfx1250 WMMA builtin signatures
// ---------------------------------------------------------------------------
typedef __attribute__((ext_vector_type(16))) __bf16        v16bf;
typedef __attribute__((ext_vector_type(8)))  float         v8f;
typedef __attribute__((ext_vector_type(4)))  unsigned int  u32x4;

#define E_DIM   1024
#define H_NUM   16
#define D_HEAD  64
#define FF_DIM  4096
#define B_SZ    2
#define N_SEQ   2048
#define ROWS    (B_SZ * N_SEQ)   // 4096 token rows
#define LN_EPS  1e-5f

enum { MODE_QKV = 0, MODE_OUTPROJ = 1, MODE_GELU = 2, MODE_FFN2 = 3 };

__device__ __forceinline__ v8f wmma_bf16f32(v16bf a, v16bf b, v8f c) {
  // D(f32,16x16) = A(bf16,16x32) * B(bf16,32x16) + C
  return __builtin_amdgcn_wmma_f32_16x16x32_bf16(
      /*neg_a=*/false, a, /*neg_b=*/false, b,
      /*c_mod=*/(short)0, c, /*reuse_a=*/false, /*reuse_b=*/false);
}

// ---------------------------------------------------------------------------
// fp32 -> bf16 cast (weights)
// ---------------------------------------------------------------------------
__global__ void cast_f32_to_bf16(const float* __restrict__ src,
                                 __bf16* __restrict__ dst, int n) {
  int i = blockIdx.x * blockDim.x + threadIdx.x;
  if (i < n) dst[i] = (__bf16)src[i];
}

// ---------------------------------------------------------------------------
// LayerNorm over E=1024, one block (8 waves) per row, bf16 output
// ---------------------------------------------------------------------------
__global__ __launch_bounds__(256) void ln_kernel(const float* __restrict__ x,
                                                 const float* __restrict__ gm,
                                                 const float* __restrict__ bt,
                                                 __bf16* __restrict__ out) {
  __shared__ float red[8];
  const int row  = blockIdx.x;
  const int tid  = threadIdx.x;
  const int lane = tid & 31;
  const int wave = tid >> 5;
  const float* xr = x + (size_t)row * E_DIM;

  float s = 0.0f;
  for (int j = tid; j < E_DIM; j += 256) s += xr[j];
  for (int off = 16; off; off >>= 1) s += __shfl_xor(s, off, 32);
  if (lane == 0) red[wave] = s;
  __syncthreads();
  float mean = 0.0f;
#pragma unroll
  for (int i = 0; i < 8; ++i) mean += red[i];
  mean *= (1.0f / (float)E_DIM);
  __syncthreads();

  float vs = 0.0f;
  for (int j = tid; j < E_DIM; j += 256) {
    float d = xr[j] - mean;
    vs += d * d;
  }
  for (int off = 16; off; off >>= 1) vs += __shfl_xor(vs, off, 32);
  if (lane == 0) red[wave] = vs;
  __syncthreads();
  float var = 0.0f;
#pragma unroll
  for (int i = 0; i < 8; ++i) var += red[i];
  var *= (1.0f / (float)E_DIM);
  const float rstd = rsqrtf(var + LN_EPS);

  for (int j = tid; j < E_DIM; j += 256) {
    float v = (xr[j] - mean) * rstd * gm[j] + bt[j];
    out[(size_t)row * E_DIM + j] = (__bf16)v;
  }
}

// ---------------------------------------------------------------------------
// Generic bf16 WMMA GEMM: C(f32) = A[M,K](bf16) * B[K,N](bf16) (+epilogue)
// Block tile: 128(M) x 64(N), 8 waves x 16-row slab of 4 WMMA tiles.
// B slab staged TRANSPOSED into LDS (double buffered); the staging global
// load is software-pipelined one full K-step ahead so the ds_store run never
// stalls on it.  All 4 B fragments (8 ds_load_b128) are issued before the 4
// WMMAs so LDS latency overlaps matrix work.
// ---------------------------------------------------------------------------
#define BTS 40  // padded bf16 stride of transposed B tile (64 rows x 32 K)

__global__ __launch_bounds__(256) void gemm_bf16_kernel(
    const __bf16* __restrict__ A, int lda,
    const __bf16* __restrict__ B, int ldb,
    int M, int N, int K, int mode,
    const float* __restrict__ bias,
    const float* __restrict__ resid,
    float* __restrict__ out_f32,
    __bf16* __restrict__ ob0, __bf16* __restrict__ ob1,
    __bf16* __restrict__ ob2) {
  __shared__ __align__(16) __bf16 ldsBT[2][64 * BTS];  // 2 x 5 KB

  const int tid    = threadIdx.x;
  const int lane   = tid & 31;
  const int wave   = tid >> 5;
  const int m      = lane & 15;   // row within A-frag / col within B,C-frag
  const int g      = lane >> 4;   // lane-half selector
  const int n_base = blockIdx.x * 64;
  const int row0   = blockIdx.y * 128 + wave * 16;

  const int s_idx = tid * 8;
  const int s_kk  = s_idx >> 6;   // 0..31 (K row within slab)
  const int s_cc  = s_idx & 63;   // col, multiple of 8
  const __bf16* Bs = B + (size_t)s_kk * ldb + n_base + s_cc;
  const size_t slab = (size_t)32 * ldb;  // elements per K-slab

  v8f acc[4] = {};
  const __bf16* Arow = A + (size_t)(row0 + m) * lda;

  const int niter = K >> 5;
  union { u32x4 u; __bf16 h[8]; } staged;
  staged.u = *(const u32x4*)Bs;  // slab 0, loaded one step ahead

  for (int it = 0; it <= niter; ++it) {
    __syncthreads();
    if (it < niter) {
      // store the pre-loaded slab into LDS (transposed)
      __bf16* dst = ldsBT[it & 1];
#pragma unroll
      for (int i = 0; i < 8; ++i) dst[(s_cc + i) * BTS + s_kk] = staged.h[i];
      // issue next slab load a full iteration early; prefetch slab+2
      if (it + 1 < niter) {
        staged.u = *(const u32x4*)(Bs + (size_t)(it + 1) * slab);
        if (it + 2 < niter)
          __builtin_prefetch(Bs + (size_t)(it + 2) * slab, 0, 0);
      }
    }
    if (it > 0) {
      const int k0 = (it - 1) * 32;
      const __bf16* bt = ldsBT[(it - 1) & 1];

      // A fragment (contiguous K, matching B): elem e -> K = g*16+e
      union { v16bf v; u32x4 u[2]; } af;
      af.u[0] = *(const u32x4*)(Arow + k0 + g * 16);
      af.u[1] = *(const u32x4*)(Arow + k0 + g * 16 + 8);

      // batch all 4 B fragments before the WMMAs
      union { v16bf v; u32x4 u[2]; } bf[4];
#pragma unroll
      for (int t = 0; t < 4; ++t) {
        const __bf16* bp = bt + (t * 16 + m) * BTS + g * 16;
        bf[t].u[0] = *(const u32x4*)bp;
        bf[t].u[1] = *(const u32x4*)(bp + 8);
      }
#pragma unroll
      for (int t = 0; t < 4; ++t)
        acc[t] = wmma_bf16f32(af.v, bf[t].v, acc[t]);
    }
  }

  // Epilogue.  C/D layout: lane elem r -> row = r + 8*g, col = lane&15.
#pragma unroll
  for (int t = 0; t < 4; ++t) {
    const int col = n_base + t * 16 + m;
#pragma unroll
    for (int r = 0; r < 8; ++r) {
      const int row = row0 + r + 8 * g;
      const float v = acc[t][r];
      if (mode == MODE_QKV) {
        const __bf16 bb = (__bf16)v;
        if (col < E_DIM) {
          ob0[(size_t)row * E_DIM + col] = bb;             // Q [rows,E]
        } else if (col < 2 * E_DIM) {
          ob1[(size_t)row * E_DIM + (col - E_DIM)] = bb;   // K [rows,E]
        } else {
          // V stored transposed per head: vt[b, h, d, n]
          const int c2 = col - 2 * E_DIM;
          const int hh = c2 >> 6, dd = c2 & 63;
          const int bb_ = row >> 11, nn = row & (N_SEQ - 1);
          ob2[((size_t)(bb_ * H_NUM + hh) * D_HEAD + dd) * N_SEQ + nn] = bb;
        }
      } else if (mode == MODE_GELU) {
        const float u  = v + bias[col];
        const float ge = 0.5f * u * (1.0f + erff(u * 0.70710678118654752f));
        ob0[(size_t)row * N + col] = (__bf16)ge;
      } else {  // MODE_OUTPROJ / MODE_FFN2: +bias +residual, f32 out
        out_f32[(size_t)row * N + col] =
            v + bias[col] + resid[(size_t)row * N + col];
      }
    }
  }
}

// ---------------------------------------------------------------------------
// Attention: one block per (b, h, 16-query tile).
// Pass1: S = (Q K^T) * D^0.25 via WMMA -> LDS f32 (16 x 2048).
// Softmax: per-row max/exp; unnormalized exp stored bf16 in LDS; 1/sum kept
// per row and folded into the output epilogue.
// Pass2: O = P V via WMMA, split-K across all 8 waves, V read from the
// pre-transposed vt[b,h,d,n] buffer (two b128 loads per fragment).
// ---------------------------------------------------------------------------
#define SSTR 2056  // f32 score stride
#define PSTR 2056  // bf16 prob stride

__global__ __launch_bounds__(256) void attn_kernel(
    const __bf16* __restrict__ Q, const __bf16* __restrict__ Km,
    const __bf16* __restrict__ Vt, __bf16* __restrict__ O) {
  __shared__ float sc[16 * SSTR];                     // ~128.5 KB
  __shared__ __align__(16) __bf16 pb[16 * PSTR];      // ~64.3 KB
  __shared__ float part[4 * 16 * 16];                 // 4 KB split-K partials
  __shared__ float invrow[16];

  const int blk  = blockIdx.x;
  const int qt   = blk & 127;          // N_SEQ/16 = 128
  const int h    = (blk >> 7) & 15;
  const int b    = blk >> 11;
  const int tid  = threadIdx.x;
  const int lane = tid & 31;
  const int wave = tid >> 5;
  const int m    = lane & 15;
  const int g    = lane >> 4;

  const size_t qrow0 = (size_t)(b * N_SEQ + qt * 16);

  // Q fragments for both K-steps (d = 0..31, 32..63), held in registers.
  const __bf16* Qp = Q + (qrow0 + m) * E_DIM + h * D_HEAD;
  union { v16bf v; u32x4 u[2]; } aq[2];
#pragma unroll
  for (int ks = 0; ks < 2; ++ks) {
    aq[ks].u[0] = *(const u32x4*)(Qp + ks * 32 + g * 16);
    aq[ks].u[1] = *(const u32x4*)(Qp + ks * 32 + g * 16 + 8);
  }

  const float sscale = 2.8284271247461903f;  // 64^0.25  (qk / scale**0.5)

  // -------- Pass 1: scores --------
  const __bf16* Kbase = Km + (size_t)b * N_SEQ * E_DIM + h * D_HEAD;
  for (int jt = wave; jt < N_SEQ / 16; jt += 8) {
    const int j0 = jt * 16;
    v8f s = {};
#pragma unroll
    for (int ks = 0; ks < 2; ++ks) {
      // B = K^T: B[d, j] = K[j, d]; 16 contiguous d per lane.
      union { v16bf v; u32x4 u[2]; } bk;
      const __bf16* kp = Kbase + (size_t)(j0 + m) * E_DIM + ks * 32 + g * 16;
      bk.u[0] = *(const u32x4*)kp;
      bk.u[1] = *(const u32x4*)(kp + 8);
      s = wmma_bf16f32(aq[ks].v, bk.v, s);
    }
#pragma unroll
    for (int r = 0; r < 8; ++r)
      sc[(r + 8 * g) * SSTR + j0 + m] = s[r] * sscale;
  }
  __syncthreads();

  // -------- Softmax: 16 rows, 2 per wave; pb = exp(s - max) in bf16 --------
#pragma unroll
  for (int rr = 0; rr < 2; ++rr) {
    const int row = wave * 2 + rr;
    const float* srow = &sc[row * SSTR];
    __bf16* prow = &pb[row * PSTR];
    float mx = -3.4e38f;
    for (int j = lane; j < N_SEQ; j += 32) mx = fmaxf(mx, srow[j]);
    for (int off = 16; off; off >>= 1) mx = fmaxf(mx, __shfl_xor(mx, off, 32));
    float sum = 0.0f;
    for (int j = lane; j < N_SEQ; j += 32) {
      float e = __expf(srow[j] - mx);
      prow[j] = (__bf16)e;
      sum += e;
    }
    for (int off = 16; off; off >>= 1) sum += __shfl_xor(sum, off, 32);
    if (lane == 0) invrow[row] = 1.0f / sum;
  }
  __syncthreads();

  // -------- Pass 2: O = P V, all 8 waves (split-K over keys) --------
  const int n0 = (wave & 3) * 16;   // d-tile
  const int jh = wave >> 2;         // key half
  v8f o = {};
  const __bf16* vrow =
      Vt + ((size_t)(b * H_NUM + h) * D_HEAD + n0 + m) * N_SEQ;
  const __bf16* prow = pb + m * PSTR;
  for (int j0 = jh * 1024; j0 < jh * 1024 + 1024; j0 += 32) {
    union { v16bf v; u32x4 u[2]; } ap, bv;
    ap.u[0] = *(const u32x4*)(prow + j0 + g * 16);
    ap.u[1] = *(const u32x4*)(prow + j0 + g * 16 + 8);
    bv.u[0] = *(const u32x4*)(vrow + j0 + g * 16);
    bv.u[1] = *(const u32x4*)(vrow + j0 + g * 16 + 8);
    o = wmma_bf16f32(ap.v, bv.v, o);
  }
  if (wave >= 4) {
#pragma unroll
    for (int r = 0; r < 8; ++r)
      part[((wave - 4) * 16 + (r + 8 * g)) * 16 + m] = o[r];
  }
  __syncthreads();
  if (wave < 4) {
#pragma unroll
    for (int r = 0; r < 8; ++r) {
      const int row = r + 8 * g;
      const float val =
          (o[r] + part[(wave * 16 + row) * 16 + m]) * invrow[row];
      O[(qrow0 + row) * E_DIM + h * D_HEAD + n0 + m] = (__bf16)val;
    }
  }
}

// ---------------------------------------------------------------------------
// Host-side orchestration
// ---------------------------------------------------------------------------
extern "C" void kernel_launch(void* const* d_in, const int* in_sizes, int n_in,
                              void* d_out, int out_size, void* d_ws,
                              size_t ws_size, hipStream_t stream) {
  const float* x     = (const float*)d_in[0];
  const float* w_qkv = (const float*)d_in[1];
  const float* w_out = (const float*)d_in[2];
  const float* b_out = (const float*)d_in[3];
  const float* ln1_g = (const float*)d_in[4];
  const float* ln1_b = (const float*)d_in[5];
  const float* ln2_g = (const float*)d_in[6];
  const float* ln2_b = (const float*)d_in[7];
  const float* w1    = (const float*)d_in[8];
  const float* b1    = (const float*)d_in[9];
  const float* w2    = (const float*)d_in[10];
  const float* b2    = (const float*)d_in[11];

  char*  ws  = (char*)d_ws;
  size_t off = 0;
  auto alloc = [&](size_t bytes) -> void* {
    void* p = ws + off;
    off = (off + bytes + 255) & ~(size_t)255;
    return p;
  };

  __bf16* wqkv_bf = (__bf16*)alloc((size_t)E_DIM * 3 * E_DIM * 2);
  __bf16* wout_bf = (__bf16*)alloc((size_t)E_DIM * E_DIM * 2);
  __bf16* w1_bf   = (__bf16*)alloc((size_t)E_DIM * FF_DIM * 2);
  __bf16* w2_bf   = (__bf16*)alloc((size_t)FF_DIM * E_DIM * 2);
  __bf16* xn_bf   = (__bf16*)alloc((size_t)ROWS * E_DIM * 2);
  __bf16* q_bf    = (__bf16*)alloc((size_t)ROWS * E_DIM * 2);
  __bf16* k_bf    = (__bf16*)alloc((size_t)ROWS * E_DIM * 2);
  __bf16* vt_bf   = (__bf16*)alloc((size_t)ROWS * E_DIM * 2);  // [B,H,D,N]
  __bf16* wavg_bf = (__bf16*)alloc((size_t)ROWS * E_DIM * 2);
  float*  x1_f    = (float*)alloc((size_t)ROWS * E_DIM * 4);
  __bf16* xn2_bf  = (__bf16*)alloc((size_t)ROWS * E_DIM * 2);
  __bf16* h_bf    = (__bf16*)alloc((size_t)ROWS * FF_DIM * 2);

  // 1) weight casts
  {
    int n;
    n = E_DIM * 3 * E_DIM;
    cast_f32_to_bf16<<<(n + 255) / 256, 256, 0, stream>>>(w_qkv, wqkv_bf, n);
    n = E_DIM * E_DIM;
    cast_f32_to_bf16<<<(n + 255) / 256, 256, 0, stream>>>(w_out, wout_bf, n);
    n = E_DIM * FF_DIM;
    cast_f32_to_bf16<<<(n + 255) / 256, 256, 0, stream>>>(w1, w1_bf, n);
    n = FF_DIM * E_DIM;
    cast_f32_to_bf16<<<(n + 255) / 256, 256, 0, stream>>>(w2, w2_bf, n);
  }

  // 2) LN1
  ln_kernel<<<ROWS, 256, 0, stream>>>(x, ln1_g, ln1_b, xn_bf);

  // 3) QKV projection (split epilogue; V written transposed per head)
  {
    dim3 grid(3 * E_DIM / 64, ROWS / 128);
    gemm_bf16_kernel<<<grid, 256, 0, stream>>>(
        xn_bf, E_DIM, wqkv_bf, 3 * E_DIM, ROWS, 3 * E_DIM, E_DIM, MODE_QKV,
        nullptr, nullptr, nullptr, q_bf, k_bf, vt_bf);
  }

  // 4) attention
  attn_kernel<<<B_SZ * H_NUM * (N_SEQ / 16), 256, 0, stream>>>(q_bf, k_bf,
                                                               vt_bf, wavg_bf);

  // 5) output projection + bias + residual -> x1 (f32)
  {
    dim3 grid(E_DIM / 64, ROWS / 128);
    gemm_bf16_kernel<<<grid, 256, 0, stream>>>(
        wavg_bf, E_DIM, wout_bf, E_DIM, ROWS, E_DIM, E_DIM, MODE_OUTPROJ,
        b_out, x, x1_f, nullptr, nullptr, nullptr);
  }

  // 6) LN2
  ln_kernel<<<ROWS, 256, 0, stream>>>(x1_f, ln2_g, ln2_b, xn2_bf);

  // 7) FFN up + exact GELU
  {
    dim3 grid(FF_DIM / 64, ROWS / 128);
    gemm_bf16_kernel<<<grid, 256, 0, stream>>>(
        xn2_bf, E_DIM, w1_bf, FF_DIM, ROWS, FF_DIM, E_DIM, MODE_GELU, b1,
        nullptr, nullptr, h_bf, nullptr, nullptr);
  }

  // 8) FFN down + bias + residual -> d_out (f32)
  {
    dim3 grid(E_DIM / 64, ROWS / 128);
    gemm_bf16_kernel<<<grid, 256, 0, stream>>>(
        h_bf, FF_DIM, w2_bf, E_DIM, ROWS, E_DIM, FF_DIM, MODE_FFN2, b2, x1_f,
        (float*)d_out, nullptr, nullptr, nullptr);
  }
}
